// MultiHeadAttentionGRUDecoder_35708358099654
// MI455X (gfx1250) — matrix operand order repaired and checked by reading, not compiled, not run on hardware
//
#include <hip/hip_runtime.h>
#include <hip/hip_bf16.h>

// MultiHeadAttention + GRU decoder for MI455X (gfx1250, wave32).
// B=32, T=64, S=512, D=512, H=8, DK=DV=64.
//
// Loop-invariant K/V projection (17 GFLOP) -> v_wmma_f32_16x16x32_f16 with
// f16 operands pre-converted in workspace; A and B fragments are both pure
// 16B vector loads (weights stored transposed f16). K-loop unrolled x4.
// Sequential 64-step recurrence -> 2 fused fp32 kernels per step; weights
// read once per step; working set (~49 MB) is L2-resident (192 MB L2).

#define BB 32
#define TT 64
#define SS 512
#define DD 512
#define HH 8
#define DKV 64

typedef __attribute__((ext_vector_type(16))) _Float16 v16h;
typedef __attribute__((ext_vector_type(8)))  _Float16 h8;
typedef __attribute__((ext_vector_type(8)))  float    v8f;

// ---------------------------------------------------------------------------
// f32 -> f16 conversion (vectorized), n must be a multiple of 4
// ---------------------------------------------------------------------------
__global__ __launch_bounds__(256) void f32_to_f16(
    const float* __restrict__ in, _Float16* __restrict__ out, int n) {
  int i = (blockIdx.x * 256 + threadIdx.x) * 4;
  if (i < n) {
    float4 v = *(const float4*)(in + i);
    out[i + 0] = (_Float16)v.x;
    out[i + 1] = (_Float16)v.y;
    out[i + 2] = (_Float16)v.z;
    out[i + 3] = (_Float16)v.w;
  }
}

// ---------------------------------------------------------------------------
// W[512,512] f32 -> Wt[512,512] f16 transposed (Wt[n][k] = W[k][n]).
// 32x33 LDS tile, coalesced loads and stores. Grid (16,16), block (32,8).
// ---------------------------------------------------------------------------
__global__ __launch_bounds__(256) void transpose_to_f16(
    const float* __restrict__ in, _Float16* __restrict__ out) {
  __shared__ float tile[32][33];
  int tx = threadIdx.x, ty = threadIdx.y;
  int n0 = blockIdx.x * 32, k0 = blockIdx.y * 32;
#pragma unroll
  for (int r = 0; r < 32; r += 8)
    tile[ty + r][tx] = in[(size_t)(k0 + ty + r) * DD + (n0 + tx)];
  __syncthreads();
#pragma unroll
  for (int r = 0; r < 32; r += 8)
    out[(size_t)(n0 + ty + r) * DD + (k0 + tx)] = (_Float16)tile[tx][ty + r];
}

// ---------------------------------------------------------------------------
// h0 = tanh(encoder_output[b, last_nonpad] @ W_he + b_he)
// ---------------------------------------------------------------------------
__global__ __launch_bounds__(256) void h0_kernel(
    const int* __restrict__ src_seq, const float* __restrict__ enc,
    const float* __restrict__ W_he, const float* __restrict__ b_he,
    float* __restrict__ h0) {
  int b = blockIdx.x;
  int tid = threadIdx.x;
  __shared__ float fin[DD];
  __shared__ int cnt_s[256];

  int c = 0;
  for (int s = tid; s < SS; s += 256) c += (src_seq[b * SS + s] != 0);
  cnt_s[tid] = c;
  __syncthreads();
  for (int off = 128; off > 0; off >>= 1) {
    if (tid < off) cnt_s[tid] += cnt_s[tid + off];
    __syncthreads();
  }
  int fi = cnt_s[0] - 1;
  if (fi < 0) fi = 0;

  for (int d = tid; d < DD; d += 256)
    fin[d] = enc[((size_t)b * SS + fi) * DD + d];
  __syncthreads();

  for (int n = tid; n < DD; n += 256) {
    float acc = b_he[n];
    for (int k = 0; k < DD; k++) acc += fin[k] * W_he[k * DD + n];
    h0[b * DD + n] = tanhf(acc);
  }
}

// ---------------------------------------------------------------------------
// K/V projection: C[16384,512] = A[16384,512] @ W[512,512] + bias, f16 WMMA.
// A f16 row-major; W as Wt[N,K] f16 (transposed) so both fragments load as
// two 16B vectors. One 16x16 tile per wave, K-loop step 32, unrolled x4.
// mode 0: store Kt[b,h,dk,s] (f16)   mode 1: store V[b,h,s,dv] (f16)
// ---------------------------------------------------------------------------
__global__ __launch_bounds__(128) void gemm_kv_wmma(
    const _Float16* __restrict__ A, const _Float16* __restrict__ Wt,
    const float* __restrict__ bias, _Float16* __restrict__ out, int mode) {
  const int Kd = 512;
  int waveId = blockIdx.x * 4 + (threadIdx.x >> 5);  // 32768 waves total
  int lane   = threadIdx.x & 31;
  int ti = waveId >> 5;        // 0..1023  (M tiles, M=16384)
  int tj = waveId & 31;        // 0..31    (N tiles, N=512)
  int i0 = ti * 16, j0 = tj * 16;
  int mrow = lane & 15;        // M (for A) / N (for B, C/D) index within tile
  int kgrp = lane >> 4;        // half-wave K-group

  // A fragment (16x32 f16): lane holds K = kgrp*8+[0,8) and 16+kgrp*8+[0,8)
  const _Float16* abase = A + ((size_t)(i0 + mrow)) * Kd + kgrp * 8;
  // B fragment (32x16 f16): lane holds K = kgrp*8+[0,8) and 16+kgrp*8+[0,8);
  // Wt[n][k] row-major -> two 8-half vectors at (j0+mrow, kk+kgrp*8) and +16
  const _Float16* wbase = Wt + ((size_t)(j0 + mrow)) * Kd + kgrp * 8;

  v8f c = {};
#pragma unroll 4
  for (int kk = 0; kk < Kd; kk += 32) {
    // Prefetch next A slice (lowers to global_prefetch_b8)
    if (kk + 32 < Kd) __builtin_prefetch(abase + kk + 32, 0, 1);

    v16h a, bf;
    h8 alo = *(const h8*)(abase + kk);
    h8 ahi = *(const h8*)(abase + kk + 16);
    h8 blo = *(const h8*)(wbase + kk);
    h8 bhi = *(const h8*)(wbase + kk + 16);
#pragma unroll
    for (int e = 0; e < 8; e++) {
      a[e]      = alo[e];
      a[8 + e]  = ahi[e];
      bf[e]     = blo[e];
      bf[8 + e] = bhi[e];
    }
    c = __builtin_amdgcn_wmma_f32_16x16x32_f16(
        false, a, false, bf, (short)0, c, false, false);
  }

  // C/D layout: VGPR r -> M = i0 + r + kgrp*8 ; N = j0 + (lane&15)
  int n  = j0 + mrow;
  float bv = bias[n];
  int h = n >> 6, j = n & 63;
#pragma unroll
  for (int r = 0; r < 8; r++) {
    int gm = i0 + r + kgrp * 8;
    int b  = gm >> 9;       // gm / S
    int s  = gm & 511;      // gm % S
    float val = c[r] + bv;
    size_t idx;
    if (mode == 0) idx = ((size_t)((b * HH + h) * DKV + j)) * SS + s;  // Kt
    else           idx = ((size_t)((b * HH + h) * SS  + s)) * DKV + j; // V
    out[idx] = (_Float16)val;
  }
}

// ---------------------------------------------------------------------------
// Per-step kernel 1: one workgroup per batch row b.
// q = h@Wq+bq -> masked scaled scores -> softmax -> ctx -> fc+residual -> LN.
// Also gathers the embedding row for this step into xcat[b][0:512];
// writes attn_vec into xcat[b][512:1024].
// ---------------------------------------------------------------------------
__global__ __launch_bounds__(1024) void attn_step(
    const float* __restrict__ h_in, const _Float16* __restrict__ Kt,
    const _Float16* __restrict__ Vh, const int* __restrict__ src_seq,
    const int* __restrict__ tgt_seq, const float* __restrict__ emb,
    const float* __restrict__ Wq, const float* __restrict__ bq,
    const float* __restrict__ Wfc, const float* __restrict__ bfc,
    const float* __restrict__ ln_g, const float* __restrict__ ln_b,
    float* __restrict__ xcat, int t) {
  int b = blockIdx.x, tid = threadIdx.x;
  __shared__ float hrow[DD];
  __shared__ float q[DD];
  __shared__ float sc[HH * SS];   // 16 KB
  __shared__ float ctx[DD];
  __shared__ float red[DD];
  __shared__ float hstat[2 * HH]; // per-head max, inv-sum

  // load h row; gather embedding row for GRU input
  if (tid < DD) {
    hrow[tid] = h_in[b * DD + tid];
    int row = tgt_seq[b * TT + t];
    xcat[b * 2 * DD + tid] = emb[(size_t)row * DD + tid];
  }
  __syncthreads();

  // q projection (coalesced over output column = tid)
  if (tid < DD) {
    float acc = bq[tid];
    for (int k = 0; k < DD; k++) acc += hrow[k] * Wq[k * DD + tid];
    q[tid] = acc;
  }
  __syncthreads();

  // scores[h][s] = scale * q[h,:]·Kt[b,h,:,s], masked where src_seq==PAD
  const float scale = 0.125f;  // 1/sqrt(64)
  for (int r = 0; r < 4; r++) {
    int v = tid + 1024 * r;          // 4096 = H*S values
    int hh = v >> 9, s = v & 511;
    const _Float16* kp = Kt + ((size_t)(b * HH + hh) * DKV) * SS + s;
    float acc = 0.f;
    for (int j = 0; j < DKV; j++) acc += q[hh * DKV + j] * (float)kp[j * SS];
    sc[v] = (src_seq[b * SS + s] == 0) ? -1e9f : acc * scale;
  }
  __syncthreads();

  // softmax per head (two-pass)
  if (tid < DD) {
    int hh = tid >> 6, l = tid & 63;
    float mx = -3.0e38f;
    for (int i = 0; i < 8; i++) mx = fmaxf(mx, sc[hh * SS + l * 8 + i]);
    red[tid] = mx;
  }
  __syncthreads();
  if (tid < HH) {
    float mx = -3.0e38f;
    for (int i = 0; i < 64; i++) mx = fmaxf(mx, red[tid * 64 + i]);
    hstat[tid] = mx;
  }
  __syncthreads();
  if (tid < DD) {
    int hh = tid >> 6, l = tid & 63;
    float mx = hstat[hh], sum = 0.f;
    for (int i = 0; i < 8; i++) {
      float e = expf(sc[hh * SS + l * 8 + i] - mx);
      sc[hh * SS + l * 8 + i] = e;
      sum += e;
    }
    red[tid] = sum;
  }
  __syncthreads();
  if (tid < HH) {
    float sum = 0.f;
    for (int i = 0; i < 64; i++) sum += red[tid * 64 + i];
    hstat[HH + tid] = 1.0f / sum;
  }
  __syncthreads();

  // ctx[h*64+j] = sum_s attn[h][s] * V[b,h,s,j]   (coalesced over j)
  if (tid < DD) {
    int hh = tid >> 6, j = tid & 63;
    float inv = hstat[HH + hh];
    const _Float16* vp = Vh + ((size_t)(b * HH + hh) * SS) * DKV + j;
    float acc = 0.f;
    for (int s = 0; s < SS; s++) acc += sc[hh * SS + s] * (float)vp[s * DKV];
    ctx[tid] = acc * inv;
  }
  __syncthreads();

  // fc + residual, then LayerNorm over the row
  float f = 0.f;
  if (tid < DD) {
    float acc = bfc[tid] + hrow[tid];
    for (int k = 0; k < DD; k++) acc += ctx[k] * Wfc[k * DD + tid];
    f = acc;
    red[tid] = acc;
  }
  __syncthreads();
  for (int off = 256; off > 0; off >>= 1) {
    if (tid < off) red[tid] += red[tid + off];
    __syncthreads();
  }
  float mu = red[0] * (1.0f / 512.0f);
  __syncthreads();
  if (tid < DD) { float d = f - mu; red[tid] = d * d; }
  __syncthreads();
  for (int off = 256; off > 0; off >>= 1) {
    if (tid < off) red[tid] += red[tid + off];
    __syncthreads();
  }
  float var = red[0] * (1.0f / 512.0f);
  if (tid < DD) {
    float y = (f - mu) * rsqrtf(var + 1e-5f) * ln_g[tid] + ln_b[tid];
    xcat[b * 2 * DD + DD + tid] = y;  // attn_vec -> second half of GRU input
  }
}

// ---------------------------------------------------------------------------
// Per-step kernel 2: GRU cell, torch gate order (r, z, n).
// Grid: 32 WGs, each owns 16 output columns d for ALL 32 batch rows, so
// W_ih / W_hh columns are read exactly once per step. Two half-threads per
// (b,d) split the k-loops; partials combined in LDS.
// ---------------------------------------------------------------------------
__global__ __launch_bounds__(1024) void gru_step(
    const float* __restrict__ h_in, const float* __restrict__ xcat,
    const float* __restrict__ W_ih, const float* __restrict__ b_ih,
    const float* __restrict__ W_hh, const float* __restrict__ b_hh,
    float* __restrict__ h_out, float* __restrict__ out, int t) {
  __shared__ float part[2][4][512];  // [half][r,z,i_n,h_n][(b,d) pair]
  int tid  = threadIdx.x;
  int half = tid >> 9;         // 0/1: which half of the k range
  int w    = tid & 511;        // (b, d_local) pair
  int b    = w >> 4;
  int dl   = w & 15;
  int d    = blockIdx.x * 16 + dl;

  const float* xr = xcat + b * (2 * DD) + half * DD;   // 512 x-values
  const float* hp = h_in + b * DD + half * 256;        // 256 h-values

  float pr = 0.f, pz = 0.f, pin = 0.f;
  for (int k = 0; k < DD; k++) {
    float xv = xr[k];
    const float* wrow = W_ih + (size_t)(half * DD + k) * 1536;
    pr  += xv * wrow[d];
    pz  += xv * wrow[DD + d];
    pin += xv * wrow[2 * DD + d];
  }
  float phr = 0.f, phz = 0.f, phn = 0.f;
  for (int k = 0; k < 256; k++) {
    float hv = hp[k];
    const float* wrow = W_hh + (size_t)(half * 256 + k) * 1536;
    phr += hv * wrow[d];
    phz += hv * wrow[DD + d];
    phn += hv * wrow[2 * DD + d];
  }
  part[half][0][w] = pr + phr;   // i_r + h_r (linear -> merge)
  part[half][1][w] = pz + phz;   // i_z + h_z
  part[half][2][w] = pin;        // i_n  (kept separate: n = tanh(i_n + r*h_n))
  part[half][3][w] = phn;        // h_n
  __syncthreads();

  if (tid < 512) {
    float rr  = part[0][0][tid] + part[1][0][tid] + b_ih[d]          + b_hh[d];
    float zz  = part[0][1][tid] + part[1][1][tid] + b_ih[DD + d]     + b_hh[DD + d];
    float inn = part[0][2][tid] + part[1][2][tid] + b_ih[2 * DD + d];
    float hnn = part[0][3][tid] + part[1][3][tid] + b_hh[2 * DD + d];
    float r = 1.0f / (1.0f + expf(-rr));
    float z = 1.0f / (1.0f + expf(-zz));
    float n = tanhf(inn + r * hnn);
    float hprev = h_in[b * DD + d];
    float hnew  = (1.0f - z) * n + z * hprev;
    h_out[b * DD + d] = hnew;
    out[((size_t)b * TT + t) * DD + d] = hnew;
  }
}

// ---------------------------------------------------------------------------
extern "C" void kernel_launch(void* const* d_in, const int* in_sizes, int n_in,
                              void* d_out, int out_size, void* d_ws, size_t ws_size,
                              hipStream_t stream) {
  const int*   tgt_seq = (const int*)d_in[0];
  // d_in[1] = tgt_pos (unused by reference math)
  const int*   src_seq = (const int*)d_in[2];
  const float* enc     = (const float*)d_in[3];
  const float* emb     = (const float*)d_in[4];
  const float* W_he    = (const float*)d_in[5];
  const float* b_he    = (const float*)d_in[6];
  const float* Wq      = (const float*)d_in[7];
  const float* bq      = (const float*)d_in[8];
  const float* Wk      = (const float*)d_in[9];
  const float* bk      = (const float*)d_in[10];
  const float* Wv      = (const float*)d_in[11];
  const float* bv      = (const float*)d_in[12];
  const float* Wfc     = (const float*)d_in[13];
  const float* bfc     = (const float*)d_in[14];
  const float* ln_g    = (const float*)d_in[15];
  const float* ln_b    = (const float*)d_in[16];
  const float* W_ih    = (const float*)d_in[17];
  const float* b_ih    = (const float*)d_in[18];
  const float* W_hh    = (const float*)d_in[19];
  const float* b_hh    = (const float*)d_in[20];
  float* out = (float*)d_out;

  // Workspace layout (~49 MB)
  char* ws = (char*)d_ws;
  _Float16* Kt   = (_Float16*)ws;                               // 16 MB [B,H,DK,S]
  _Float16* Vh   = (_Float16*)(ws + (size_t)16 * 1024 * 1024);  // 16 MB [B,H,S,DV]
  _Float16* encH = (_Float16*)(ws + (size_t)32 * 1024 * 1024);  // 16 MB [B*S, D] f16
  _Float16* Wt   = (_Float16*)(ws + (size_t)48 * 1024 * 1024);  // 512 KB (reused K->V)
  float* hA   = (float*)(ws + (size_t)49 * 1024 * 1024);        // 64 KB
  float* hB   = hA + BB * DD;                                   // 64 KB
  float* xcat = hB + BB * DD;                                   // 128 KB [B][2D]

  const int encN = BB * SS * DD;   // 8,388,608

  // Loop-invariant precompute
  h0_kernel<<<BB, 256, 0, stream>>>(src_seq, enc, W_he, b_he, hA);
  f32_to_f16<<<encN / 1024, 256, 0, stream>>>(enc, encH, encN);
  {
    dim3 tg(16, 16), tb(32, 8);
    transpose_to_f16<<<tg, tb, 0, stream>>>(Wk, Wt);
    gemm_kv_wmma<<<8192, 128, 0, stream>>>(encH, Wt, bk, Kt, 0);
    transpose_to_f16<<<tg, tb, 0, stream>>>(Wv, Wt);
    gemm_kv_wmma<<<8192, 128, 0, stream>>>(encH, Wt, bv, Vh, 1);
  }

  // Sequential decode: 2 kernels per step, h ping-pongs in ws
  float* hc = hA;
  float* hn = hB;
  for (int t = 0; t < TT; t++) {
    attn_step<<<BB, 1024, 0, stream>>>(hc, Kt, Vh, src_seq, tgt_seq, emb,
                                       Wq, bq, Wfc, bfc, ln_g, ln_b, xcat, t);
    gru_step<<<BB, 1024, 0, stream>>>(hc, xcat, W_ih, b_ih, W_hh, b_hh,
                                      hn, out, t);
    float* tmp = hc; hc = hn; hn = tmp;
  }
}
